// Network_33887291966161
// MI455X (gfx1250) — compile-verified, matrix operand order checked by reading, not run on hardware
//
#include <hip/hip_runtime.h>
#include <hip/hip_bf16.h>

// ---------------------------------------------------------------------------
// Types for WMMA fragments (gfx1250, wave32)
// ---------------------------------------------------------------------------
typedef __attribute__((ext_vector_type(16))) __bf16          v16bf;
typedef __attribute__((ext_vector_type(8)))  float           v8f;
typedef __attribute__((ext_vector_type(8)))  unsigned short  v8us;

union Frag {
    v8us           h[2];
    unsigned short u[16];
    v16bf          v;
};

__device__ __forceinline__ unsigned short bfbits(float x) {
    union { float f; unsigned u; } c; c.f = x;
    unsigned u = c.u;
    unsigned r = (u + 0x7FFFu + ((u >> 16) & 1u)) >> 16;   // round-to-nearest-even
    return (unsigned short)r;
}

// Problem constants
#define NB   1024          // batch
#define NA   81            // areas
#define NAP  96            // areas padded to 3 k-steps of 32
#define NP   7             // players
#define NF   32            // filters
#define NC   64            // 2F channels
#define R2   7168          // B * P rows in the player stage

// ---------------------------------------------------------------------------
// Prep kernels
// ---------------------------------------------------------------------------

// w[n][k][f] (f32) -> wt[n][f][kpad] (bf16, zero padded K)
__global__ void k_prep_w(const float* __restrict__ src, unsigned short* __restrict__ dst,
                         int nareas, int K, int Kpad) {
    int idx = blockIdx.x * blockDim.x + threadIdx.x;
    int tot = nareas * NF * Kpad;
    if (idx >= tot) return;
    int n = idx / (NF * Kpad);
    int r = idx - n * (NF * Kpad);
    int f = r / Kpad;
    int k = r - f * Kpad;
    float v = (k < K) ? src[(n * K + k) * NF + f] : 0.0f;
    dst[idx] = bfbits(v);
}

// adjacency [81][81] f32 -> [96][96] bf16 zero padded (symmetric input)
__global__ void k_prep_adj(const float* __restrict__ adj, unsigned short* __restrict__ dst) {
    int idx = blockIdx.x * blockDim.x + threadIdx.x;
    if (idx >= NAP * NAP) return;
    int i = idx / NAP, j = idx - i * NAP;
    float v = (i < NA && j < NA) ? adj[i * NA + j] : 0.0f;
    dst[idx] = bfbits(v);
}

// player_emb [7][16] f32 -> [7][32] bf16 (zero padded to a full 32-wide K-block)
__global__ void k_prep_emb(const float* __restrict__ pemb, unsigned short* __restrict__ dst) {
    int idx = blockIdx.x * blockDim.x + threadIdx.x;
    if (idx >= NP * 32) return;
    int p = idx / 32, k = idx - p * 32;
    float v = (k < 16) ? pemb[p * 16 + k] : 0.0f;
    dst[idx] = bfbits(v);
}

// build x0 bf16 [B][A][64]: [state(35) | season_emb(16) | builds(1) | zeros(12)]
__global__ void k_build_x0(const float* __restrict__ state, const int* __restrict__ season,
                           const int* __restrict__ builds, const float* __restrict__ semb,
                           unsigned short* __restrict__ dst) {
    int idx = blockIdx.x * blockDim.x + threadIdx.x;
    if (idx >= NB * NA * 64) return;
    int b = idx / (NA * 64);
    int r = idx - b * (NA * 64);
    int a = r / 64;
    int k = r - a * 64;
    float v;
    if (k < 35)       v = state[(b * NA + a) * 35 + k];
    else if (k < 51)  v = semb[season[b] * 16 + (k - 35)];
    else if (k == 51) v = (float)builds[b];
    else              v = 0.0f;
    dst[idx] = bfbits(v);
}

// ---------------------------------------------------------------------------
// Kernel A: per-area message GEMM, two M-tiles (32 rows) per wave so each
// weight fragment feeds two WMMAs.
//   messages[r, n, :] = x[r, n, :K] @ w[n]        (bf16 WMMA, f32 accum)
// Writes: h[r][n][32+f] (f32, second half of concat) and msgT[f][r][n] (bf16).
// pmode: player-encoder concat [rep | player_emb | 0] fused into the A loads
// (k-block 2 reads the prepacked bf16 embedding table).
// ---------------------------------------------------------------------------
__global__ __launch_bounds__(32)
void k_gemm_msg(const unsigned short* __restrict__ x, const unsigned short* __restrict__ wt,
                const unsigned short* __restrict__ pembbf, float* __restrict__ h,
                unsigned short* __restrict__ msgT, int rows, int ksteps, int kx, int kw,
                int pmode) {
    const int r0   = blockIdx.x * 32;     // two 16-row tiles
    const int n    = blockIdx.y;
    const int lane = threadIdx.x;
    const int ml   = lane & 15;           // M row (A) / N col (B,D) within tile
    const int kh   = lane >> 4;           // K-half selector

    const int grow0 = r0 + ml;
    const int grow1 = r0 + 16 + ml;
    const unsigned short* xrow0;
    const unsigned short* xrow1;
    int p0 = 0, p1 = 0;
    if (pmode) {
        int b0 = grow0 / NP; p0 = grow0 - b0 * NP;
        int b1 = grow1 / NP; p1 = grow1 - b1 * NP;
        xrow0 = x + ((size_t)b0 * NA + n) * kx;
        xrow1 = x + ((size_t)b1 * NA + n) * kx;
    } else {
        xrow0 = x + ((size_t)grow0 * NA + n) * kx;
        xrow1 = x + ((size_t)grow1 * NA + n) * kx;
    }

    v8f acc00 = {0.f,0.f,0.f,0.f,0.f,0.f,0.f,0.f};   // tile0 x ftile0
    v8f acc01 = acc00;                                // tile0 x ftile1
    v8f acc10 = acc00;                                // tile1 x ftile0
    v8f acc11 = acc00;                                // tile1 x ftile1

    for (int s = 0; s < ksteps; ++s) {
        const int k0 = s * 32;
        const unsigned short* pa0;
        const unsigned short* pa1;
        if (pmode && k0 >= kx) {          // embedding K-block: [emb(16) | zeros(16)]
            pa0 = pembbf + p0 * 32;
            pa1 = pembbf + p1 * 32;
        } else {
            pa0 = xrow0 + k0;
            pa1 = xrow1 + k0;
        }
        Frag fa0, fa1;
        fa0.h[0] = *(const v8us*)(pa0 + 8 * kh);
        fa0.h[1] = *(const v8us*)(pa0 + 16 + 8 * kh);
        fa1.h[0] = *(const v8us*)(pa1 + 8 * kh);
        fa1.h[1] = *(const v8us*)(pa1 + 16 + 8 * kh);

        const unsigned short* w0 = wt + ((size_t)n * NF + 0 * 16 + ml) * kw + k0 + 8 * kh;
        const unsigned short* w1 = wt + ((size_t)n * NF + 1 * 16 + ml) * kw + k0 + 8 * kh;
        Frag fb0, fb1;
        fb0.h[0] = *(const v8us*)(w0);      fb0.h[1] = *(const v8us*)(w0 + 16);
        fb1.h[0] = *(const v8us*)(w1);      fb1.h[1] = *(const v8us*)(w1 + 16);

        acc00 = __builtin_amdgcn_wmma_f32_16x16x32_bf16(false, fa0.v, false, fb0.v,
                                                        (short)0, acc00, false, false);
        acc01 = __builtin_amdgcn_wmma_f32_16x16x32_bf16(false, fa0.v, false, fb1.v,
                                                        (short)0, acc01, false, false);
        acc10 = __builtin_amdgcn_wmma_f32_16x16x32_bf16(false, fa1.v, false, fb0.v,
                                                        (short)0, acc10, false, false);
        acc11 = __builtin_amdgcn_wmma_f32_16x16x32_bf16(false, fa1.v, false, fb1.v,
                                                        (short)0, acc11, false, false);
    }

    // D layout: lane -> N col = ml (+16*ftile); VGPR v -> M row v + 8*kh
    #pragma unroll
    for (int v = 0; v < 8; ++v) {
        const int f0 = ml, f1 = 16 + ml;
        {
            const int mrow = r0 + v + 8 * kh;
            size_t hb = ((size_t)mrow * NA + n) * NC + 32;
            h[hb + f0] = acc00[v];
            h[hb + f1] = acc01[v];
            msgT[((size_t)f0 * rows + mrow) * NAP + n] = bfbits(acc00[v]);
            msgT[((size_t)f1 * rows + mrow) * NAP + n] = bfbits(acc01[v]);
        }
        {
            const int mrow = r0 + 16 + v + 8 * kh;
            size_t hb = ((size_t)mrow * NA + n) * NC + 32;
            h[hb + f0] = acc10[v];
            h[hb + f1] = acc11[v];
            msgT[((size_t)f0 * rows + mrow) * NAP + n] = bfbits(acc10[v]);
            msgT[((size_t)f1 * rows + mrow) * NAP + n] = bfbits(acc11[v]);
        }
    }
}

// ---------------------------------------------------------------------------
// Kernel B: adjacency aggregation as WMMA GEMM (adjacency is symmetric):
//   agg[r, n] = sum_a msgT[f][r][a] * adj[n][a]   per channel f.
//   A = msgT rows (M=16 rows, K=96 areas), B = adj rows (N=96 out-areas).
// Writes h[r][n][f] (first half of concat).
// ---------------------------------------------------------------------------
__global__ __launch_bounds__(32)
void k_agg(const unsigned short* __restrict__ msgT, const unsigned short* __restrict__ adjb,
           float* __restrict__ h, int rows) {
    const int r0   = blockIdx.x * 16;
    const int f    = blockIdx.y;
    const int lane = threadIdx.x;
    const int ml   = lane & 15;
    const int kh   = lane >> 4;

    const unsigned short* arow = msgT + ((size_t)f * rows + (r0 + ml)) * NAP;

    v8f acc[6];
    #pragma unroll
    for (int t = 0; t < 6; ++t) acc[t] = (v8f){0.f,0.f,0.f,0.f,0.f,0.f,0.f,0.f};

    #pragma unroll
    for (int s = 0; s < 3; ++s) {
        const int k0 = s * 32;
        Frag fa;
        fa.h[0] = *(const v8us*)(arow + k0 + 8 * kh);
        fa.h[1] = *(const v8us*)(arow + k0 + 16 + 8 * kh);
        #pragma unroll
        for (int t = 0; t < 6; ++t) {
            const unsigned short* brow = adjb + (size_t)(t * 16 + ml) * NAP + k0 + 8 * kh;
            Frag fb;
            fb.h[0] = *(const v8us*)(brow);
            fb.h[1] = *(const v8us*)(brow + 16);
            acc[t] = __builtin_amdgcn_wmma_f32_16x16x32_bf16(false, fa.v, false, fb.v,
                                                             (short)0, acc[t], false, false);
        }
    }

    #pragma unroll
    for (int t = 0; t < 6; ++t) {
        const int area = t * 16 + ml;
        if (area < NA) {
            #pragma unroll
            for (int v = 0; v < 8; ++v) {
                const int mrow = r0 + v + 8 * kh;
                h[((size_t)mrow * NA + area) * NC + f] = acc[t][v];
            }
        }
    }
}

// ---------------------------------------------------------------------------
// BatchNorm stats (two-pass): zero -> atomic partial sums -> apply
// ---------------------------------------------------------------------------
__global__ void k_zero_stats(float* stats) { stats[threadIdx.x] = 0.0f; }

__global__ void k_stats(const float* __restrict__ h, float* __restrict__ stats, int n) {
    __shared__ float ssum[256];
    __shared__ float ssq[256];
    const int tid = threadIdx.x;
    const int stride = gridDim.x * 256;
    float s = 0.f, q = 0.f;
    for (int i = blockIdx.x * 256 + tid; i < n; i += stride) {
        float v = h[i];
        s += v; q += v * v;
    }
    ssum[tid] = s; ssq[tid] = q;
    __syncthreads();
    if (tid < 64) {   // stride is a multiple of 64 -> tid%64 == channel
        float S = ssum[tid] + ssum[tid + 64] + ssum[tid + 128] + ssum[tid + 192];
        float Q = ssq[tid]  + ssq[tid + 64]  + ssq[tid + 128]  + ssq[tid + 192];
        atomicAdd(&stats[tid], S);
        atomicAdd(&stats[64 + tid], Q);
    }
}

// normalize + (relu) + (residual); optional f32 rep store, bf16 store, f32 out store
__global__ void k_bn(const float* __restrict__ h, const float* __restrict__ stats,
                     const float* __restrict__ scale, const float* __restrict__ offset,
                     float* rep, unsigned short* xout, float* fout,
                     int n, float invcnt, int dorelu, int dores) {
    int idx = blockIdx.x * blockDim.x + threadIdx.x;
    if (idx >= n) return;
    int c = idx & 63;
    float mean = stats[c] * invcnt;
    float var  = stats[64 + c] * invcnt - mean * mean;
    float y = scale[c] * (h[idx] - mean) * rsqrtf(var + 1e-5f) + offset[c];
    if (dorelu) y = fmaxf(y, 0.0f);
    if (dores)  y += rep[idx];
    if (rep)  rep[idx] = y;
    if (xout) xout[idx] = bfbits(y);
    if (fout) fout[idx] = y;
}

// ---------------------------------------------------------------------------
// Host side
// ---------------------------------------------------------------------------
static inline size_t alg(size_t x) { return (x + 255) & ~(size_t)255; }

extern "C" void kernel_launch(void* const* d_in, const int* in_sizes, int n_in,
                              void* d_out, int out_size, void* d_ws, size_t ws_size,
                              hipStream_t stream) {
    const float* state      = (const float*)d_in[0];
    const int*   season     = (const int*)  d_in[1];
    const int*   builds     = (const int*)  d_in[2];
    const float* adjacency  = (const float*)d_in[3];
    const float* season_emb = (const float*)d_in[4];
    const float* player_emb = (const float*)d_in[5];
    const float* w_enc_s    = (const float*)d_in[6];
    const float* sc_enc_s   = (const float*)d_in[7];
    const float* of_enc_s   = (const float*)d_in[8];
    const float* w_core_s   = (const float*)d_in[9];
    const float* sc_core_s  = (const float*)d_in[10];
    const float* of_core_s  = (const float*)d_in[11];
    const float* w_enc_p    = (const float*)d_in[12];
    const float* sc_enc_p   = (const float*)d_in[13];
    const float* of_enc_p   = (const float*)d_in[14];
    const float* w_core_p   = (const float*)d_in[15];
    const float* sc_core_p  = (const float*)d_in[16];
    const float* of_core_p  = (const float*)d_in[17];
    const float* sc_fin     = (const float*)d_in[18];
    const float* of_fin     = (const float*)d_in[19];
    float* out = (float*)d_out;

    char* ws = (char*)d_ws;
    size_t o = 0;
    unsigned short* adjb      = (unsigned short*)(ws + o); o = alg(o + (size_t)NAP*NAP*2);
    unsigned short* pembbf    = (unsigned short*)(ws + o); o = alg(o + (size_t)NP*32*2);
    unsigned short* wt_enc_s  = (unsigned short*)(ws + o); o = alg(o + (size_t)NA*NF*64*2);
    unsigned short* wt_core_s = (unsigned short*)(ws + o); o = alg(o + (size_t)8*NA*NF*64*2);
    unsigned short* wt_enc_p  = (unsigned short*)(ws + o); o = alg(o + (size_t)NA*NF*96*2);
    unsigned short* wt_core_p = (unsigned short*)(ws + o); o = alg(o + (size_t)8*NA*NF*64*2);
    unsigned short* x0bf      = (unsigned short*)(ws + o); o = alg(o + (size_t)NB*NA*64*2);
    unsigned short* xbf_s     = (unsigned short*)(ws + o); o = alg(o + (size_t)NB*NA*64*2);
    float*          rep_s     = (float*)(ws + o);          o = alg(o + (size_t)NB*NA*64*4);
    unsigned short* xbf_p     = (unsigned short*)(ws + o); o = alg(o + (size_t)R2*NA*64*2);
    float*          rep_p     = (float*)(ws + o);          o = alg(o + (size_t)R2*NA*64*4);
    float*          hbuf      = (float*)(ws + o);          o = alg(o + (size_t)R2*NA*64*4);
    unsigned short* msgT      = (unsigned short*)(ws + o); o = alg(o + (size_t)NF*R2*NAP*2);
    float*          stats     = (float*)(ws + o);          o = alg(o + 128*4);

    // ---- prep ----
    {
        int t = NAP * NAP;
        k_prep_adj<<<(t + 255) / 256, 256, 0, stream>>>(adjacency, adjb);
    }
    {
        int t = NP * 32;
        k_prep_emb<<<1, 256, 0, stream>>>(player_emb, pembbf);
    }
    {
        int t = NA * NF * 64;
        k_prep_w<<<(t + 255) / 256, 256, 0, stream>>>(w_enc_s, wt_enc_s, NA, 52, 64);
    }
    {
        int t = 8 * NA * NF * 64;
        k_prep_w<<<(t + 255) / 256, 256, 0, stream>>>(w_core_s, wt_core_s, 8 * NA, 64, 64);
        k_prep_w<<<(t + 255) / 256, 256, 0, stream>>>(w_core_p, wt_core_p, 8 * NA, 64, 64);
    }
    {
        int t = NA * NF * 96;
        k_prep_w<<<(t + 255) / 256, 256, 0, stream>>>(w_enc_p, wt_enc_p, NA, 80, 96);
    }
    {
        int t = NB * NA * 64;
        k_build_x0<<<(t + 255) / 256, 256, 0, stream>>>(state, season, builds, season_emb, x0bf);
    }

    auto run_layer = [&](const unsigned short* xin, int kx, const unsigned short* wt, int kw,
                         int ks, int pmode, int rows, const float* sc, const float* of,
                         int dores, float* rep, unsigned short* xout) {
        dim3 ga(rows / 32, NA);
        k_gemm_msg<<<ga, 32, 0, stream>>>(xin, wt, pembbf, hbuf, msgT,
                                          rows, ks, kx, kw, pmode);
        dim3 gb(rows / 16, NF);
        k_agg<<<gb, 32, 0, stream>>>(msgT, adjb, hbuf, rows);
        k_zero_stats<<<1, 128, 0, stream>>>(stats);
        int n = rows * NA * NC;
        k_stats<<<2048, 256, 0, stream>>>(hbuf, stats, n);
        k_bn<<<(n + 255) / 256, 256, 0, stream>>>(hbuf, stats, sc, of, rep, xout, nullptr,
                                                  n, 1.0f / (float)(rows * NA), 1, dores);
    };

    // shared encoder + 8 shared cores (rows = 1024)
    run_layer(x0bf, 64, wt_enc_s, 64, 2, 0, NB, sc_enc_s, of_enc_s, 0, rep_s, xbf_s);
    for (int l = 0; l < 8; ++l)
        run_layer(xbf_s, 64, wt_core_s + (size_t)l * NA * NF * 64, 64, 2, 0, NB,
                  sc_core_s + l * 64, of_core_s + l * 64, 1, rep_s, xbf_s);

    // player encoder (concat fused; rows = 7168) + 8 player cores
    run_layer(xbf_s, 64, wt_enc_p, 96, 3, 1, R2, sc_enc_p, of_enc_p, 0, rep_p, xbf_p);
    for (int l = 0; l < 8; ++l)
        run_layer(xbf_p, 64, wt_core_p + (size_t)l * NA * NF * 64, 64, 2, 0, R2,
                  sc_core_p + l * 64, of_core_p + l * 64, 1, rep_p, xbf_p);

    // final batch norm over rep_p -> d_out [B, P, A, 2F]
    {
        int n = R2 * NA * NC;
        k_zero_stats<<<1, 128, 0, stream>>>(stats);
        k_stats<<<2048, 256, 0, stream>>>(rep_p, stats, n);
        k_bn<<<(n + 255) / 256, 256, 0, stream>>>(rep_p, stats, sc_fin, of_fin,
                                                  nullptr, nullptr, out,
                                                  n, 1.0f / (float)(R2 * NA), 0, 0);
    }
    (void)in_sizes; (void)n_in; (void)out_size; (void)ws_size;
}
